// SelfAttentionBlock_17540646437671
// MI455X (gfx1250) — matrix operand order repaired and checked by reading, not compiled, not run on hardware
//
#include <hip/hip_runtime.h>

// ---------- types ----------
typedef __attribute__((ext_vector_type(16))) __bf16 v16bf;
typedef __attribute__((ext_vector_type(8)))  float  v8f;
typedef unsigned short u16;
typedef unsigned int   u32;

__device__ __forceinline__ u16 f2bf(float f) {
    union { float f; u32 u; } c; c.f = f;
    u32 r = c.u + 0x7FFFu + ((c.u >> 16) & 1u);
    return (u16)(r >> 16);
}

__device__ __forceinline__ u32 pack2bf(float lo, float hi) {
    return (u32)f2bf(lo) | ((u32)f2bf(hi) << 16);
}

__device__ __forceinline__ v8f vzero() {
    v8f z; for (int e = 0; e < 8; ++e) z[e] = 0.f; return z;
}

__device__ __forceinline__ v16bf make_v16(uint4 lo, uint4 hi) {
    union { uint4 q[2]; v16bf v; } u; u.q[0] = lo; u.q[1] = hi; return u.v;
}

// A fragment (16x32 bf16, MxK): lane<16 holds row=lane, K = k0..k0+7 (v0-3) and
// k0+16..k0+23 (v4-7); lanes>=16 hold K+8 / K+24.  Source is row-major [M][K].
__device__ __forceinline__ v16bf load_a(const u16* p, int ld, int row, int k0, int lane) {
    const u16* b = p + (size_t)(row + (lane & 15)) * ld + k0 + ((lane >> 4) << 3);
    uint4 lo = *(const uint4*)(b);
    uint4 hi = *(const uint4*)(b + 16);
    return make_v16(lo, hi);
}

// B fragment (32x16 bf16, KxN): lane<16 holds col=lane, K = k0..k0+7 (v0-3) and
// k0+16..k0+23 (v4-7); lanes>=16 hold K+8 / K+24.  Source stored transposed, row-major [N][K] (stride ld).
__device__ __forceinline__ v16bf load_b(const u16* p, int ld, int k0, int col, int lane) {
    const u16* b = p + (size_t)(col + (lane & 15)) * ld + k0 + ((lane >> 4) << 3);
    uint4 lo = *(const uint4*)(b);
    uint4 hi = *(const uint4*)(b + 16);
    return make_v16(lo, hi);
}

__device__ __forceinline__ v8f wmma_bf16(v16bf a, v16bf b, v8f c) {
    return __builtin_amdgcn_wmma_f32_16x16x32_bf16(false, a, false, b, (short)0, c, false, false);
}

// ---------- kernel 1: fp32 -> bf16 weight convert ----------
__global__ void cvt_kernel(const float* __restrict__ w, u16* __restrict__ o, int n) {
    int i = blockIdx.x * blockDim.x + threadIdx.x;
    if (i < n) o[i] = f2bf(w[i]);
}

// ---------- kernel 2: GroupNorm -> hT [b][n][c] bf16 ----------
// One block per (b, group). Group = 16 channels x 1024 pixels.
// Each thread owns all 16 channels of 4 pixels -> packed b128 output stores.
__global__ void __launch_bounds__(256) gn_kernel(const float* __restrict__ x,
                                                 const float* __restrict__ gamma,
                                                 const float* __restrict__ beta,
                                                 u16* __restrict__ hT) {
    int bg = blockIdx.x, b = bg >> 5, g = bg & 31;
    const float* xp = x + (size_t)(b * 512 + g * 16) * 1024;
    int t = threadIdx.x;

    float vals[64], s = 0.f, ss = 0.f;
    for (int c = 0; c < 16; ++c)
        for (int nn = 0; nn < 4; ++nn) {
            float v = xp[c * 1024 + nn * 256 + t];   // coalesced
            vals[c * 4 + nn] = v; s += v; ss += v * v;
        }

    __shared__ float red[2][256];
    red[0][t] = s; red[1][t] = ss; __syncthreads();
    for (int off = 128; off > 0; off >>= 1) {
        if (t < off) { red[0][t] += red[0][t + off]; red[1][t] += red[1][t + off]; }
        __syncthreads();
    }
    float mean = red[0][0] * (1.f / 16384.f);
    float var  = red[1][0] * (1.f / 16384.f) - mean * mean;
    float inv  = rsqrtf(var + 1e-5f);

    float ga[16], be[16];
    for (int c = 0; c < 16; ++c) { ga[c] = gamma[g * 16 + c] * inv; be[c] = beta[g * 16 + c]; }

    for (int nn = 0; nn < 4; ++nn) {
        int n = nn * 256 + t;
        float y[16];
        for (int c = 0; c < 16; ++c) y[c] = (vals[c * 4 + nn] - mean) * ga[c] + be[c];
        uint4 w0, w1;
        w0.x = pack2bf(y[0], y[1]);   w0.y = pack2bf(y[2], y[3]);
        w0.z = pack2bf(y[4], y[5]);   w0.w = pack2bf(y[6], y[7]);
        w1.x = pack2bf(y[8], y[9]);   w1.y = pack2bf(y[10], y[11]);
        w1.z = pack2bf(y[12], y[13]); w1.w = pack2bf(y[14], y[15]);
        u16* dst = hT + ((size_t)(b * 1024 + n)) * 512 + g * 16;
        *(uint4*)(dst) = w0;
        *(uint4*)(dst + 8) = w1;
    }
}

// ---------- kernel 3: QKV GEMM  qkv[o][n] = qkv_w[o][:] . h[:, n] + bias ----------
// Block = 8 waves, each wave a 64x64 tile; block tile 128(o) x 256(n).
__global__ void __launch_bounds__(256) qkv_kernel(const u16* __restrict__ wq,   // [1536][512]
                                                  const u16* __restrict__ hT,   // [B][1024][512]
                                                  const float* __restrict__ bias,
                                                  u16* __restrict__ qb,         // [B*8][1024][64]
                                                  u16* __restrict__ kb,         // [B*8][1024][64]
                                                  u16* __restrict__ vb) {       // [B*8][64][1024]
    int b = blockIdx.z;
    int wid = threadIdx.x >> 5, lane = threadIdx.x & 31;
    int o0 = blockIdx.x * 128 + (wid >> 2) * 64;
    int n0 = blockIdx.y * 256 + (wid & 3) * 64;
    const u16* hb = hT + (size_t)b * 1024 * 512;

    v8f acc[4][4];
    for (int i = 0; i < 4; ++i) for (int j = 0; j < 4; ++j) acc[i][j] = vzero();

    for (int k = 0; k < 512; k += 32) {
        v16bf af[4], bfr[4];
        for (int i = 0; i < 4; ++i) af[i]  = load_a(wq, 512, o0 + i * 16, k, lane);
        for (int j = 0; j < 4; ++j) bfr[j] = load_b(hb, 512, k, n0 + j * 16, lane);
        for (int i = 0; i < 4; ++i)
            for (int j = 0; j < 4; ++j)
                acc[i][j] = wmma_bf16(af[i], bfr[j], acc[i][j]);
    }

    // Epilogue. o-tile (64 wide) lies inside one 512-band and one 64-head band,
    // so qkv-select and head are wave-uniform.
    int half = lane >> 4, lc = lane & 15;
    int sel = o0 >> 9;                 // 0=q, 1=k, 2=v (uniform)
    int hh  = (o0 >> 6) & 7;           // head (uniform)
    size_t bh = (size_t)(b * 8 + hh);
    int dbase = (o0 & 63) + half * 8;  // == half*8 (o0 is 64-aligned)

    float bv[8];  // bias for this lane's 8 d-values of tile-row block i
    if (sel < 2) {
        // q/k layout [bh][n][d]: lane's 8 elements are d-consecutive -> one b128 store
        u16* dst = (sel == 0 ? qb : kb) + bh * 1024 * 64;
        for (int i = 0; i < 4; ++i) {
            for (int e = 0; e < 8; ++e) bv[e] = bias[o0 + i * 16 + half * 8 + e];
            for (int j = 0; j < 4; ++j) {
                int n = n0 + j * 16 + lc;
                uint4 w;
                w.x = pack2bf(acc[i][j][0] + bv[0], acc[i][j][1] + bv[1]);
                w.y = pack2bf(acc[i][j][2] + bv[2], acc[i][j][3] + bv[3]);
                w.z = pack2bf(acc[i][j][4] + bv[4], acc[i][j][5] + bv[5]);
                w.w = pack2bf(acc[i][j][6] + bv[6], acc[i][j][7] + bv[7]);
                *(uint4*)(dst + (size_t)n * 64 + dbase + i * 16) = w;
            }
        }
    } else {
        // v layout [bh][d][n]: n varies across lanes -> coalesced b16 stores
        u16* dst = vb + bh * 64 * 1024;
        for (int i = 0; i < 4; ++i) {
            for (int e = 0; e < 8; ++e) bv[e] = bias[o0 + i * 16 + half * 8 + e];
            for (int e = 0; e < 8; ++e) {
                u16* rowp = dst + (size_t)(i * 16 + dbase + e) * 1024 + n0 + lc;
                for (int j = 0; j < 4; ++j)
                    rowp[j * 16] = f2bf(acc[i][j][e] + bv[e]);
            }
        }
    }
}

// ---------- kernel 4: flash attention, one wave per 32 query rows ----------
__global__ void __launch_bounds__(128) attn_kernel(const u16* __restrict__ qb,
                                                   const u16* __restrict__ kb,
                                                   const u16* __restrict__ vb,
                                                   u16* __restrict__ aout) { // [b][n][c]
    __shared__ u16 pshare[4][32 * 64];
    int wid = threadIdx.x >> 5, lane = threadIdx.x & 31;
    int gw = blockIdx.x * 4 + wid;       // 4096 waves total
    int bh = gw >> 5;                    // (b, head), 0..127
    int n0 = (gw & 31) * 32;
    int b = bh >> 3, h = bh & 7;
    const u16* qp = qb + (size_t)bh * 1024 * 64;
    const u16* kp = kb + (size_t)bh * 1024 * 64;
    const u16* vp = vb + (size_t)bh * 64 * 1024;
    u16* ps = &pshare[wid][0];
    const float scale = 0.125f; // 64^-0.5

    // Q A-fragments: 2 row tiles x 2 k-steps (d = 0..63), reused across all column tiles
    v16bf aq[2][2];
    for (int rt = 0; rt < 2; ++rt)
        for (int ks = 0; ks < 2; ++ks)
            aq[rt][ks] = load_a(qp, 64, n0 + rt * 16, ks * 32, lane);

    v8f o_acc[2][4];
    float mrow[2][8], lrow[2][8];
    for (int rt = 0; rt < 2; ++rt) {
        for (int dt = 0; dt < 4; ++dt) o_acc[rt][dt] = vzero();
        for (int i = 0; i < 8; ++i) { mrow[rt][i] = -1e30f; lrow[rt][i] = 0.f; }
    }

    for (int mt = 0; mt < 16; ++mt) {
        int m0 = mt * 64;
        // ---- S = scale * Q^T K  (32 x 64) ----
        v8f s_acc[2][4];
        for (int rt = 0; rt < 2; ++rt) for (int ct = 0; ct < 4; ++ct) s_acc[rt][ct] = vzero();
        for (int ks = 0; ks < 2; ++ks) {
            v16bf bk[4];
            for (int ct = 0; ct < 4; ++ct) bk[ct] = load_b(kp, 64, ks * 32, m0 + ct * 16, lane);
            for (int rt = 0; rt < 2; ++rt)
                for (int ct = 0; ct < 4; ++ct)
                    s_acc[rt][ct] = wmma_bf16(aq[rt][ks], bk[ct], s_acc[rt][ct]);
        }
        // ---- online softmax; lane holds rows {i + 8*(lane>=16)} for both S and O ----
        for (int rt = 0; rt < 2; ++rt) {
            float rmax[8], rsum[8];
            for (int i = 0; i < 8; ++i) {
                float v = s_acc[rt][0][i];
                for (int ct = 1; ct < 4; ++ct) v = fmaxf(v, s_acc[rt][ct][i]);
                rmax[i] = v * scale;
            }
            for (int off = 1; off < 16; off <<= 1)
                for (int i = 0; i < 8; ++i)
                    rmax[i] = fmaxf(rmax[i], __shfl_xor(rmax[i], off, 32));
            for (int i = 0; i < 8; ++i) {
                float mn = fmaxf(mrow[rt][i], rmax[i]);
                float alpha = __expf(mrow[rt][i] - mn);
                mrow[rt][i] = mn;
                lrow[rt][i] *= alpha;
                for (int dt = 0; dt < 4; ++dt) o_acc[rt][dt][i] *= alpha;
                rsum[i] = 0.f;
            }
            for (int ct = 0; ct < 4; ++ct)
                for (int i = 0; i < 8; ++i) {
                    float p = __expf(s_acc[rt][ct][i] * scale - mrow[rt][i]);
                    rsum[i] += p;
                    int row = rt * 16 + i + (lane >> 4) * 8;
                    int col = ct * 16 + (lane & 15);
                    ps[row * 64 + col] = f2bf(p);   // C-layout -> A-layout via LDS
                }
            for (int off = 1; off < 16; off <<= 1)
                for (int i = 0; i < 8; ++i)
                    rsum[i] += __shfl_xor(rsum[i], off, 32);
            for (int i = 0; i < 8; ++i) lrow[rt][i] += rsum[i];
        }
        asm volatile("s_wait_dscnt 0" ::: "memory");
        // ---- O += P V  (K dim = m, 64 this tile) ----
        for (int ks = 0; ks < 2; ++ks) {
            v16bf ap[2], bvf[4];
            for (int rt = 0; rt < 2; ++rt) ap[rt] = load_a(ps, 64, rt * 16, ks * 32, lane);
            for (int dt = 0; dt < 4; ++dt) bvf[dt] = load_b(vp, 1024, m0 + ks * 32, dt * 16, lane);
            for (int rt = 0; rt < 2; ++rt)
                for (int dt = 0; dt < 4; ++dt)
                    o_acc[rt][dt] = wmma_bf16(ap[rt], bvf[dt], o_acc[rt][dt]);
        }
    }

    // ---- finalize: O / l, write [b][n][h*64+d] bf16 (coalesced across lanes) ----
    int half = lane >> 4, lc = lane & 15;
    for (int rt = 0; rt < 2; ++rt)
        for (int i = 0; i < 8; ++i) {
            float invl = 1.f / lrow[rt][i];
            int n = n0 + rt * 16 + i + half * 8;
            u16* rowp = aout + ((size_t)(b * 1024) + n) * 512 + h * 64 + lc;
            for (int dt = 0; dt < 4; ++dt)
                rowp[dt * 16] = f2bf(o_acc[rt][dt][i] * invl);
        }
}

// ---------- kernel 5: proj GEMM + bias + residual (fp32 out) ----------
__global__ void __launch_bounds__(256) proj_kernel(const u16* __restrict__ wp,  // [512][512]
                                                   const u16* __restrict__ aT,  // [B][1024][512]
                                                   const float* __restrict__ bias,
                                                   const float* __restrict__ x,
                                                   float* __restrict__ out) {
    int b = blockIdx.z;
    int wid = threadIdx.x >> 5, lane = threadIdx.x & 31;
    int o0 = blockIdx.x * 128 + (wid >> 2) * 64;
    int n0 = blockIdx.y * 256 + (wid & 3) * 64;
    const u16* ab = aT + (size_t)b * 1024 * 512;

    v8f acc[4][4];
    for (int i = 0; i < 4; ++i) for (int j = 0; j < 4; ++j) acc[i][j] = vzero();

    for (int k = 0; k < 512; k += 32) {
        v16bf af[4], bfr[4];
        for (int i = 0; i < 4; ++i) af[i]  = load_a(wp, 512, o0 + i * 16, k, lane);
        for (int j = 0; j < 4; ++j) bfr[j] = load_b(ab, 512, k, n0 + j * 16, lane);
        for (int i = 0; i < 4; ++i)
            for (int j = 0; j < 4; ++j)
                acc[i][j] = wmma_bf16(af[i], bfr[j], acc[i][j]);
    }

    int half = lane >> 4, lc = lane & 15;
    for (int i = 0; i < 4; ++i) {
        int orow = o0 + i * 16 + half * 8;
        float bv[8];
        for (int e = 0; e < 8; ++e) bv[e] = bias[orow + e];
        for (int j = 0; j < 4; ++j) {
            size_t base = ((size_t)(b * 512 + orow)) * 1024 + n0 + j * 16 + lc;
            for (int e = 0; e < 8; ++e)
                out[base + (size_t)e * 1024] = x[base + (size_t)e * 1024] + acc[i][j][e] + bv[e];
        }
    }
}

// ---------- host launcher ----------
extern "C" void kernel_launch(void* const* d_in, const int* in_sizes, int n_in,
                              void* d_out, int out_size, void* d_ws, size_t ws_size,
                              hipStream_t stream) {
    const float* x        = (const float*)d_in[0];
    const float* gn_gamma = (const float*)d_in[1];
    const float* gn_beta  = (const float*)d_in[2];
    const float* qkv_w    = (const float*)d_in[3];
    const float* qkv_b    = (const float*)d_in[4];
    const float* proj_w   = (const float*)d_in[5];
    const float* proj_b   = (const float*)d_in[6];
    float* out = (float*)d_out;

    size_t off = 0;
    auto carve = [&](size_t bytes) {
        void* p = (char*)d_ws + off;
        off += (bytes + 255) & ~(size_t)255;
        return p;
    };
    u16* wq_bf = (u16*)carve((size_t)1536 * 512 * 2);          // 1.5 MB
    u16* wp_bf = (u16*)carve((size_t)512 * 512 * 2);           // 0.5 MB
    u16* hT    = (u16*)carve((size_t)16 * 1024 * 512 * 2);     // 16 MB
    u16* qb    = (u16*)carve((size_t)128 * 1024 * 64 * 2);     // 16 MB
    u16* kb    = (u16*)carve((size_t)128 * 1024 * 64 * 2);     // 16 MB
    u16* vb    = (u16*)carve((size_t)128 * 64 * 1024 * 2);     // 16 MB
    u16* aT    = hT;  // attention output reuses hT (fully consumed by then)

    cvt_kernel<<<(1536 * 512 + 255) / 256, 256, 0, stream>>>(qkv_w, wq_bf, 1536 * 512);
    cvt_kernel<<<(512 * 512 + 255) / 256, 256, 0, stream>>>(proj_w, wp_bf, 512 * 512);

    gn_kernel<<<16 * 32, 256, 0, stream>>>(x, gn_gamma, gn_beta, hT);

    qkv_kernel<<<dim3(1536 / 128, 1024 / 256, 16), 256, 0, stream>>>(wq_bf, hT, qkv_b, qb, kb, vb);

    attn_kernel<<<4096 / 4, 128, 0, stream>>>(qb, kb, vb, aT);

    proj_kernel<<<dim3(512 / 128, 1024 / 256, 16), 256, 0, stream>>>(wp_bf, aT, proj_b, x, out);
}
